// ProgressiveTrack_7473243095569
// MI455X (gfx1250) — matrix operand order repaired and checked by reading, not compiled, not run on hardware
//
#include <hip/hip_runtime.h>
#include <math.h>

#define BB 32
#define NT1 64
#define NT2 128
#define CF 256
#define NPROP 64

typedef __attribute__((ext_vector_type(16))) _Float16 v16h;
typedef __attribute__((ext_vector_type(8)))  float    v8f;

__device__ inline v8f wmma_f16(v16h a, v16h b, v8f c) {
  // D = A(16x32) * B(32x16) + C, f32 accumulate
  return __builtin_amdgcn_wmma_f32_16x16x32_f16(false, a, false, b, (short)0, c, false, false);
}

// ---------------------------------------------------------------------------
// Stage 1: cosine cross-correlation  cos[b,m,n] = tf.sf / max(|tf||sf|,1e-8)
// ---------------------------------------------------------------------------
__global__ void k_cos(const float* __restrict__ tf, const float* __restrict__ sf,
                      float* __restrict__ cosb) {
  __shared__ float tn[NT1];
  int b = blockIdx.x, n = threadIdx.x;
  if (n < NT1) {
    float s = 0.f;
    for (int c = 0; c < CF; ++c) { float v = tf[((size_t)b*CF + c)*NT1 + n]; s += v*v; }
    tn[n] = sqrtf(s);
  }
  float sn = 0.f;
  for (int c = 0; c < CF; ++c) { float v = sf[((size_t)b*CF + c)*NT2 + n]; sn += v*v; }
  sn = sqrtf(sn);
  __syncthreads();
  for (int m = 0; m < NT1; ++m) {
    float d = 0.f;
    for (int c = 0; c < CF; ++c)
      d += tf[((size_t)b*CF + c)*NT1 + m] * sf[((size_t)b*CF + c)*NT2 + n];
    cosb[((size_t)b*NT1 + m)*NT2 + n] = d / fmaxf(tn[m]*sn, 1e-8f);
  }
}

// ---------------------------------------------------------------------------
// Stage 2: pre1[b,o,m] = W1[o,1:260] . [txyz[b,m,:]; tf[b,:,m]]
// ---------------------------------------------------------------------------
__global__ void k_pre1(const float* __restrict__ W1, const float* __restrict__ txyz,
                       const float* __restrict__ tf, float* __restrict__ pre1) {
  int b = blockIdx.x, m = blockIdx.y, o = threadIdx.x;
  const float* wr = W1 + (size_t)o*260;
  float acc = wr[1]*txyz[((size_t)b*NT1+m)*3+0]
            + wr[2]*txyz[((size_t)b*NT1+m)*3+1]
            + wr[3]*txyz[((size_t)b*NT1+m)*3+2];
  for (int c = 0; c < CF; ++c) acc += wr[4+c]*tf[((size_t)b*CF+c)*NT1+m];
  pre1[((size_t)b*CF+o)*NT1+m] = acc;
}

// ---------------------------------------------------------------------------
// Stage 3 (heavy, fused): mlp layers 2,3 with WMMA + max over template dim.
// One block per batch. 16 waves; wave w owns output rows [16w,16w+16).
// W2/W3 staged in LDS pre-swizzled to A-fragment order; h tile ping-pongs
// through a 64-column swizzled B-fragment buffer.
// ---------------------------------------------------------------------------
__global__ void __launch_bounds__(512) k_mlpmax(
    const float* __restrict__ cosb, const float* __restrict__ pre1,
    const float* __restrict__ W1, const float* __restrict__ s1, const float* __restrict__ b1,
    const float* __restrict__ W2, const float* __restrict__ s2, const float* __restrict__ b2,
    const float* __restrict__ W3, const float* __restrict__ s3, const float* __restrict__ b3,
    float* __restrict__ fus0) {
  extern __shared__ char smem[];
  _Float16* w2s = (_Float16*)smem;     // 65536 halves (8kt x 16ot x 32lane x 16e)
  _Float16* w3s = w2s + 65536;         // 65536 halves
  _Float16* hb  = w3s + 65536;         // 16384 halves (8kt x 4nt x 32lane x 16e)
  float* a1  = (float*)(hb + 16384);   // 256
  float* c1m = a1 + 256;               // 256
  float* s2a = c1m + 256; float* b2a = s2a + 256;
  float* s3a = b2a + 256; float* b3a = s3a + 256;

  int b = blockIdx.x;
  int tid = threadIdx.x;
  int wv = tid >> 5, lane = tid & 31;

  // stage weights in A-fragment swizzle
  for (int idx = tid; idx < 65536; idx += 512) {
    int e = idx & 15, ln = (idx >> 4) & 31, ot = (idx >> 9) & 15, kt = idx >> 13;
    int mr = ln & 15, kb = (ln >> 4) * 16;
    int k = kt*32 + kb + e;
    int row = ot*16 + mr;
    w2s[idx] = (_Float16)W2[(size_t)row*CF + k];
    w3s[idx] = (_Float16)W3[(size_t)row*CF + k];
  }
  float rs1 = 0.f, rb1 = 0.f;
  if (tid < 256) {
    rs1 = s1[tid]; rb1 = b1[tid];
    a1[tid] = rs1 * W1[(size_t)tid*260];   // s1 * W1[:,0] (cos channel)
    s2a[tid] = s2[tid]; b2a[tid] = b2[tid];
    s3a[tid] = s3[tid]; b3a[tid] = b3[tid];
  }
  __syncthreads();

  for (int nc = 0; nc < 2; ++nc) {
    int ncol0 = nc*64;
    v8f mx[4];
    #pragma unroll
    for (int nt = 0; nt < 4; ++nt) { v8f z = {}; mx[nt] = z; }

    for (int m = 0; m < NT1; ++m) {
      if (tid < 256)
        c1m[tid] = rs1 * pre1[((size_t)b*CF + tid)*NT1 + m] + rb1;
      __syncthreads();

      // h1 = relu(a1*cos + c1m) -> swizzled B-fragments (this thread fills e=wv)
      {
        int nl_lo = lane & 15, khalf = (lane >> 4) * 16;
        #pragma unroll
        for (int nt = 0; nt < 4; ++nt) {
          int n = ncol0 + nt*16 + nl_lo;
          float cv = cosb[((size_t)b*NT1 + m)*NT2 + n];
          #pragma unroll
          for (int kt = 0; kt < 8; ++kt) {
            int k = kt*32 + khalf + wv;
            float h = fmaxf(a1[k]*cv + c1m[k], 0.f);
            hb[(((kt<<2) + nt)*32 + lane)*16 + wv] = (_Float16)h;
          }
        }
      }
      __syncthreads();

      // GEMM1: h2 = W2 * h1
      v8f c1[4];
      #pragma unroll
      for (int nt = 0; nt < 4; ++nt) { v8f z = {}; c1[nt] = z; }
      for (int kt = 0; kt < 8; ++kt) {
        v16h af = *(const v16h*)&w2s[(((kt<<4) + wv)*32 + lane)*16];
        #pragma unroll
        for (int nt = 0; nt < 4; ++nt) {
          v16h bf = *(const v16h*)&hb[(((kt<<2) + nt)*32 + lane)*16];
          c1[nt] = wmma_f16(af, bf, c1[nt]);
        }
      }
      __syncthreads();

      // epilogue1 -> swizzled h2 back into hb
      {
        int nl_lo = lane & 15;
        #pragma unroll
        for (int nt = 0; nt < 4; ++nt) {
          #pragma unroll
          for (int v = 0; v < 8; ++v) {
            int o = wv*16 + (lane>>4)*8 + v;
            float y = fmaxf(s2a[o]*c1[nt][v] + b2a[o], 0.f);
            int kt2 = o >> 5, e2 = o & 15, ln2 = ((o >> 4) & 1)*16 + nl_lo;
            hb[(((kt2<<2) + nt)*32 + ln2)*16 + e2] = (_Float16)y;
          }
        }
      }
      __syncthreads();

      // GEMM2: h3 = W3 * h2 ; relu ; running max over m
      v8f c2[4];
      #pragma unroll
      for (int nt = 0; nt < 4; ++nt) { v8f z = {}; c2[nt] = z; }
      for (int kt = 0; kt < 8; ++kt) {
        v16h af = *(const v16h*)&w3s[(((kt<<4) + wv)*32 + lane)*16];
        #pragma unroll
        for (int nt = 0; nt < 4; ++nt) {
          v16h bf = *(const v16h*)&hb[(((kt<<2) + nt)*32 + lane)*16];
          c2[nt] = wmma_f16(af, bf, c2[nt]);
        }
      }
      #pragma unroll
      for (int nt = 0; nt < 4; ++nt)
        #pragma unroll
        for (int v = 0; v < 8; ++v) {
          int o = wv*16 + (lane>>4)*8 + v;
          float y = fmaxf(s3a[o]*c2[nt][v] + b3a[o], 0.f);
          mx[nt][v] = fmaxf(mx[nt][v], y);
        }
    } // m

    #pragma unroll
    for (int nt = 0; nt < 4; ++nt)
      #pragma unroll
      for (int v = 0; v < 8; ++v) {
        int o = wv*16 + (lane>>4)*8 + v;
        int n = ncol0 + nt*16 + (lane & 15);
        fus0[((size_t)b*CF + o)*NT2 + n] = mx[nt][v];
      }
  } // nc
}

// ---------------------------------------------------------------------------
// Generic WMMA MLP layer: Y[b,o,n] = act(s[o]*(W[o,:C].X[b,:,n]) + bias[o])
// X stored [B][Cstore][N]; grid (B, ceil(O/64), ceil(N/128)); block 128 (4 waves)
// ---------------------------------------------------------------------------
__global__ void __launch_bounds__(128) k_mlp(
    const float* __restrict__ X, const float* __restrict__ W,
    const float* __restrict__ sc, const float* __restrict__ bi,
    float* __restrict__ Y, int C, int Cstore, int O, int N, int act) {
  extern __shared__ char smem[];
  _Float16* Xs = (_Float16*)smem;
  int b = blockIdx.x;
  int tid = threadIdx.x, lane = tid & 31, wv = tid >> 5;
  int orow0 = blockIdx.y*64 + wv*16;
  int ncol0 = blockIdx.z*128;
  int ncols = N - ncol0; if (ncols > 128) ncols = 128;
  int KT = (C + 31) >> 5;

  // stage B operand (X) as swizzled f16 fragments, zero-padded
  for (int idx = tid; idx < KT*32*128; idx += 128) {
    int k = idx >> 7, n = idx & 127;
    float v = 0.f;
    if (k < C && n < ncols) v = X[((size_t)b*Cstore + k)*N + ncol0 + n];
    int kt = k >> 5, e = k & 15, ln = ((k >> 4) & 1)*16 + (n & 15), nt = n >> 4;
    Xs[(((kt<<3) + nt)*32 + ln)*16 + e] = (_Float16)v;
  }
  __syncthreads();

  int NTt = (ncols + 15) >> 4;
  v8f acc[8];
  #pragma unroll
  for (int nt = 0; nt < 8; ++nt) { v8f z = {}; acc[nt] = z; }
  int mrow = orow0 + (lane & 15);
  int kb = (lane >> 4)*16;

  for (int kt = 0; kt < KT; ++kt) {
    v16h af;
    #pragma unroll
    for (int e = 0; e < 16; ++e) {
      int k = kt*32 + kb + e;
      float w = 0.f;
      if (mrow < O && k < C) w = W[(size_t)mrow*C + k];
      af[e] = (_Float16)w;
    }
    #pragma unroll
    for (int nt = 0; nt < 8; ++nt)
      if (nt < NTt) {
        v16h bf = *(const v16h*)&Xs[(((kt<<3) + nt)*32 + lane)*16];
        acc[nt] = wmma_f16(af, bf, acc[nt]);
      }
  }
  #pragma unroll
  for (int nt = 0; nt < 8; ++nt)
    if (nt < NTt)
      #pragma unroll
      for (int v = 0; v < 8; ++v) {
        int o = orow0 + (lane >> 4)*8 + v;
        int n = ncol0 + nt*16 + (lane & 15);
        if (o < O && n < N) {
          float y = sc[o]*acc[nt][v] + bi[o];
          if (act) y = fmaxf(y, 0.f);
          Y[((size_t)b*O + o)*N + n] = y;
        }
      }
}

// ---------------------------------------------------------------------------
// fx = [search_xyz^T ; fus]
// ---------------------------------------------------------------------------
__global__ void k_fx(const float* __restrict__ sxyz, const float* __restrict__ fus,
                     float* __restrict__ fx) {
  int b = blockIdx.x, n = threadIdx.x;
  for (int c = 0; c < 3; ++c)
    fx[((size_t)b*259 + c)*NT2 + n] = sxyz[((size_t)b*NT2 + n)*3 + c];
  for (int c = 0; c < CF; ++c)
    fx[((size_t)b*259 + 3 + c)*NT2 + n] = fus[((size_t)b*CF + c)*NT2 + n];
}

// vote = fx + v3 ; vote_xyz, vfeat(=[sigmoid(cla); vote[3:]]), cla out
__global__ void k_votefin(const float* __restrict__ fx, const float* __restrict__ v3,
                          const float* __restrict__ clav, float* __restrict__ vxyz,
                          float* __restrict__ vfeat, float* __restrict__ outv,
                          float* __restrict__ outcl) {
  int b = blockIdx.x, n = threadIdx.x;
  for (int c = 0; c < 259; ++c) {
    float v = fx[((size_t)b*259 + c)*NT2 + n] + v3[((size_t)b*259 + c)*NT2 + n];
    if (c < 3) {
      vxyz[((size_t)b*NT2 + n)*3 + c] = v;
      outv[((size_t)b*NT2 + n)*3 + c] = v;
    } else {
      vfeat[((size_t)b*257 + (c - 2))*NT2 + n] = v;
    }
  }
  float cl = clav[(size_t)b*NT2 + n];
  vfeat[((size_t)b*257)*NT2 + n] = 1.f/(1.f + expf(-cl));
  outcl[(size_t)b*NT2 + n] = cl;
}

// ---------------------------------------------------------------------------
// Farthest point sampling (sequential, per-batch block) + centers
// ---------------------------------------------------------------------------
__global__ void k_fps(const float* __restrict__ vxyz, int* __restrict__ fidx,
                      float* __restrict__ cent, float* __restrict__ outcent) {
  __shared__ float rv[NT2];
  __shared__ int   ri[NT2];
  __shared__ int   sel[NPROP];
  int b = blockIdx.x, n = threadIdx.x;
  float px = vxyz[((size_t)b*NT2+n)*3+0];
  float py = vxyz[((size_t)b*NT2+n)*3+1];
  float pz = vxyz[((size_t)b*NT2+n)*3+2];
  float dst = 1e10f;
  int last = 0;
  if (n == 0) sel[0] = 0;
  for (int it = 1; it < NPROP; ++it) {
    float lx = vxyz[((size_t)b*NT2+last)*3+0];
    float ly = vxyz[((size_t)b*NT2+last)*3+1];
    float lz = vxyz[((size_t)b*NT2+last)*3+2];
    float dx = px-lx, dy = py-ly, dz = pz-lz;
    dst = fminf(dst, dx*dx + dy*dy + dz*dz);
    rv[n] = dst; ri[n] = n;
    __syncthreads();
    for (int s = NT2/2; s > 0; s >>= 1) {
      if (n < s) {
        float v2 = rv[n+s]; int i2 = ri[n+s];
        if (v2 > rv[n] || (v2 == rv[n] && i2 < ri[n])) { rv[n] = v2; ri[n] = i2; }
      }
      __syncthreads();
    }
    last = ri[0];
    if (n == 0) sel[it] = last;
    __syncthreads();
  }
  if (n < NPROP) {
    int s = sel[n];
    fidx[(size_t)b*NPROP + n] = s;
    for (int c = 0; c < 3; ++c) {
      float v = vxyz[((size_t)b*NT2 + s)*3 + c];
      cent[((size_t)b*NPROP + n)*3 + c] = v;
      outcent[((size_t)b*NPROP + n)*3 + c] = v;
    }
  }
}

// Ball query: first 16 in-radius indices ascending; pad with first index
__global__ void k_ball(const float* __restrict__ cent, const float* __restrict__ vxyz,
                       int* __restrict__ gidx) {
  int b = blockIdx.x, s = threadIdx.x;
  float cx = cent[((size_t)b*NPROP+s)*3+0];
  float cy = cent[((size_t)b*NPROP+s)*3+1];
  float cz = cent[((size_t)b*NPROP+s)*3+2];
  int cnt = 0, first = 0; bool have = false;
  int* g = gidx + ((size_t)b*NPROP + s)*16;
  for (int n = 0; n < NT2; ++n) {
    float dx = vxyz[((size_t)b*NT2+n)*3+0]-cx;
    float dy = vxyz[((size_t)b*NT2+n)*3+1]-cy;
    float dz = vxyz[((size_t)b*NT2+n)*3+2]-cz;
    if (dx*dx+dy*dy+dz*dz < 0.09f) {
      if (!have) { first = n; have = true; }
      if (cnt < 16) g[cnt] = n;
      ++cnt;
    }
  }
  for (int j = cnt; j < 16; ++j) g[j] = first;
}

// Group: sa_in[b, c, s*16+j] = [gxyz(3); vfeat gathered(257)]
__global__ void k_group(const float* __restrict__ vxyz, const float* __restrict__ cent,
                        const float* __restrict__ vfeat, const int* __restrict__ gidx,
                        float* __restrict__ sain) {
  int b = blockIdx.x, c = blockIdx.y;
  for (int col = threadIdx.x; col < NPROP*16; col += blockDim.x) {
    int s = col >> 4, j = col & 15;
    int gp = gidx[((size_t)b*NPROP + s)*16 + j];
    float v;
    if (c < 3) v = vxyz[((size_t)b*NT2 + gp)*3 + c] - cent[((size_t)b*NPROP + s)*3 + c];
    else       v = vfeat[((size_t)b*257 + (c-3))*NT2 + gp];
    sain[((size_t)b*260 + c)*1024 + col] = v;
  }
}

__global__ void k_max16(const float* __restrict__ x, float* __restrict__ pf) {
  int b = blockIdx.x, o = blockIdx.y, s = threadIdx.x;
  const float* r = x + ((size_t)b*CF + o)*1024 + (size_t)s*16;
  float m = r[0];
  for (int j = 1; j < 16; ++j) m = fmaxf(m, r[j]);
  pf[((size_t)b*CF + o)*NPROP + s] = m;
}

__global__ void k_boxes(const float* __restrict__ off, const float* __restrict__ cent,
                        float* __restrict__ outb) {
  int b = blockIdx.x, p = threadIdx.x;
  for (int c = 0; c < 3; ++c)
    outb[((size_t)b*NPROP + p)*4 + c] =
        off[((size_t)b*4 + c)*NPROP + p] + cent[((size_t)b*NPROP + p)*3 + c];
  outb[((size_t)b*NPROP + p)*4 + 3] = off[((size_t)b*4 + 3)*NPROP + p];
}

// ---------------------------------------------------------------------------
extern "C" void kernel_launch(void* const* d_in, const int* in_sizes, int n_in,
                              void* d_out, int out_size, void* d_ws, size_t ws_size,
                              hipStream_t stream) {
  (void)n_in; (void)out_size; (void)ws_size;
  const float* sxyz = (const float*)d_in[0];
  const float* sf   = (const float*)d_in[1];
  const float* txyz = (const float*)d_in[2];
  const float* tf   = (const float*)d_in[3];

  struct Lay { const float *W, *s, *b; };
  Lay mlp[3], fea[2], cla[3], vote[3], sa[3], prop[3];
  int i = 4;
  auto take = [&](Lay* arr, int nl, bool wsb) {
    for (int j = 0; j < nl; ++j) {
      arr[j].W = (const float*)d_in[i++];
      if (wsb) { arr[j].s = (const float*)d_in[i++]; arr[j].b = (const float*)d_in[i++]; }
      else     { arr[j].b = (const float*)d_in[i++]; arr[j].s = (const float*)d_in[i++]; }
    }
  };
  if (in_sizes[4] == 260*256) {
    // top-level insertion order, per-layer W,s,b
    take(mlp,3,true); take(fea,2,true); take(cla,3,true);
    take(vote,3,true); take(sa,3,true); take(prop,3,true);
  } else {
    // tree-sorted keys: cla,fea,mlp,prop,sa,vote ; per-layer W,b,s
    take(cla,3,false); take(fea,2,false); take(mlp,3,false);
    take(prop,3,false); take(sa,3,false); take(vote,3,false);
  }

  // workspace layout (f32 elements)
  float* p = (float*)d_ws;
  float* cosb = p; p += (size_t)BB*NT1*NT2;
  float* pre1 = p; p += (size_t)BB*CF*NT1;
  float* fus0 = p; p += (size_t)BB*CF*NT2;
  float* fus1 = p; p += (size_t)BB*CF*NT2;
  float* fus2 = p; p += (size_t)BB*CF*NT2;
  float* cAb  = p; p += (size_t)BB*CF*NT2;
  float* cBb  = p; p += (size_t)BB*CF*NT2;
  float* clav = p; p += (size_t)BB*NT2;
  float* fxb  = p; p += (size_t)BB*259*NT2;
  float* v1b  = p; p += (size_t)BB*CF*NT2;
  float* v2b  = p; p += (size_t)BB*CF*NT2;
  float* v3b  = p; p += (size_t)BB*259*NT2;
  float* vxyz = p; p += (size_t)BB*NT2*3;
  float* vft  = p; p += (size_t)BB*257*NT2;
  float* cent = p; p += (size_t)BB*NPROP*3;
  float* pfb  = p; p += (size_t)BB*CF*NPROP;
  float* pAb  = p; p += (size_t)BB*CF*NPROP;
  float* pBb  = p; p += (size_t)BB*CF*NPROP;
  float* offb = p; p += (size_t)BB*4*NPROP;
  float* sain = p; p += (size_t)BB*260*1024;
  float* saA  = p; p += (size_t)BB*CF*1024;
  float* saB  = p; p += (size_t)BB*CF*1024;
  int* fidx = (int*)p; p += (size_t)BB*NPROP;
  int* gidx = (int*)p; p += (size_t)BB*NPROP*16;

  float* out    = (float*)d_out;
  float* out_v  = out;            // vote_xyz [B,128,3]
  float* out_b  = out + 12288;    // boxes    [B,64,4]
  float* out_c  = out + 20480;    // centers  [B,64,3]
  float* out_cl = out + 26624;    // cla      [B,1,128]

  auto L = [&](const float* X, const Lay& l, float* Y, int C, int Cs, int O, int N, int act) {
    dim3 g(BB, (unsigned)((O + 63)/64), (unsigned)((N + 127)/128));
    int KT = (C + 31)/32;
    k_mlp<<<g, dim3(128), (size_t)KT*8192, stream>>>(X, l.W, l.s, l.b, Y, C, Cs, O, N, act);
  };

  // xcorr + layer-1 factorization
  k_cos<<<dim3(BB), dim3(128), 0, stream>>>(tf, sf, cosb);
  k_pre1<<<dim3(BB, NT1), dim3(256), 0, stream>>>(mlp[0].W, txyz, tf, pre1);

  // fused mlp layers 2,3 + max over template points (WMMA heavy kernel)
  size_t shmm = (size_t)(65536 + 65536 + 16384)*2 + 6*256*4; // 301056 B
  k_mlpmax<<<dim3(BB), dim3(512), shmm, stream>>>(
      cosb, pre1,
      mlp[0].W, mlp[0].s, mlp[0].b,
      mlp[1].W, mlp[1].s, mlp[1].b,
      mlp[2].W, mlp[2].s, mlp[2].b, fus0);

  // fea (last no act)
  L(fus0, fea[0], fus1, 256, 256, 256, NT2, 1);
  L(fus1, fea[1], fus2, 256, 256, 256, NT2, 0);
  // cla
  L(fus2, cla[0], cAb, 256, 256, 256, NT2, 1);
  L(cAb,  cla[1], cBb, 256, 256, 256, NT2, 1);
  L(cBb,  cla[2], clav, 256, 256, 1,  NT2, 0);
  // vote (residual MLP)
  k_fx<<<dim3(BB), dim3(128), 0, stream>>>(sxyz, fus2, fxb);
  L(fxb, vote[0], v1b, 259, 259, 256, NT2, 1);
  L(v1b, vote[1], v2b, 256, 256, 256, NT2, 1);
  L(v2b, vote[2], v3b, 256, 256, 259, NT2, 0);
  k_votefin<<<dim3(BB), dim3(128), 0, stream>>>(fxb, v3b, clav, vxyz, vft, out_v, out_cl);

  // vote aggregation: FPS + ball query + grouping
  k_fps<<<dim3(BB), dim3(128), 0, stream>>>(vxyz, fidx, cent, out_c);
  k_ball<<<dim3(BB), dim3(NPROP), 0, stream>>>(cent, vxyz, gidx);
  k_group<<<dim3(BB, 260), dim3(256), 0, stream>>>(vxyz, cent, vft, gidx, sain);

  // SA MLP (all relu) + max over samples
  L(sain, sa[0], saA, 260, 260, 256, 1024, 1);
  L(saA,  sa[1], saB, 256, 256, 256, 1024, 1);
  L(saB,  sa[2], saA, 256, 256, 256, 1024, 1);
  k_max16<<<dim3(BB, CF), dim3(NPROP), 0, stream>>>(saA, pfb);

  // proposals
  L(pfb, prop[0], pAb, 256, 256, 256, NPROP, 1);
  L(pAb, prop[1], pBb, 256, 256, 256, NPROP, 1);
  L(pBb, prop[2], offb, 256, 256, 4,  NPROP, 0);
  k_boxes<<<dim3(BB), dim3(NPROP), 0, stream>>>(offb, cent, out_b);
}
